// CountHistogram_2319282340172
// MI455X (gfx1250) — compile-verified, matrix operand order checked by reading, not run on hardware
//
#include <hip/hip_runtime.h>

// CountHistogram: B=128, C=4, Q=32, D=2048, NBINS=30
// hist[b,c,q,bin] = sum_d [bin(simmat[b,c,q,d])==bin] * (dtoks[b,d]!=-1) * (qtoks[b,q]!=-1)

#define NBINS 30
#define BB 128
#define CC 4
#define QQ 32
#define DD 2048
#define LSTRIDE 33              // padded per-lane LDS stride in words (bank-conflict avoidance)
#define WAVES_PER_BLOCK 8

typedef __attribute__((ext_vector_type(16))) _Float16 v16h;
typedef __attribute__((ext_vector_type(8)))  float    v8f;

__global__ __launch_bounds__(256) void CountHistogram_kernel(
    const float* __restrict__ simmat,
    const int*   __restrict__ dtoks,
    const int*   __restrict__ qtoks,
    float*       __restrict__ out)
{
    // 8 waves x 32 lanes x 33 words = 33,792 B
    __shared__ unsigned cnt[WAVES_PER_BLOCK * 32 * LSTRIDE];

    const int tid  = threadIdx.x;
    const int wave = tid >> 5;
    const int lane = tid & 31;
    const int row  = blockIdx.x * WAVES_PER_BLOCK + wave;   // [0, B*C*Q)
    const int b    = row >> 7;                              // row / (C*Q), C*Q=128
    const int q    = row & (QQ - 1);                        // row % Q

    // ---- zero this lane's private histogram slot (incl. padding words 30..32,
    //      which feed zeros into the second WMMA's N=30/31 pad columns) ----
    unsigned* my = &cnt[(wave * 32 + lane) * LSTRIDE];
    #pragma unroll
    for (int i = 0; i < LSTRIDE; ++i) my[i] = 0u;
    __syncthreads();

    const int      qtok = qtoks[b * QQ + q];
    const unsigned qw   = (qtok != -1) ? 1u : 0u;

    const float4* sp = reinterpret_cast<const float4*>(simmat + (size_t)row * DD);
    const int4*   tp = reinterpret_cast<const int4*>(dtoks  + (size_t)b   * DD);

    // ---- main streaming loop: 16 iters x (float4 + int4) per lane = 2048 elems/row ----
    #pragma unroll 4
    for (int it = 0; it < DD / (32 * 4); ++it) {
        const int v = it * 32 + lane;                        // float4 index, coalesced
        __builtin_prefetch((const void*)(sp + v + 64), 0, 1); // global_prefetch_b8, ~2 iters ahead
        const float4 s = sp[v];
        const int4   t = tp[v];

        // exact reference FP order: ((x + 1.00001f) * 0.5f) * 29.0f, truncate
        {
            const int bin = (int)(((s.x + 1.00001f) * 0.5f) * 29.0f);
            atomicAdd(&my[bin], (t.x != -1) ? qw : 0u);      // ds_add_u32 (no rtn)
        }
        {
            const int bin = (int)(((s.y + 1.00001f) * 0.5f) * 29.0f);
            atomicAdd(&my[bin], (t.y != -1) ? qw : 0u);
        }
        {
            const int bin = (int)(((s.z + 1.00001f) * 0.5f) * 29.0f);
            atomicAdd(&my[bin], (t.z != -1) ? qw : 0u);
        }
        {
            const int bin = (int)(((s.w + 1.00001f) * 0.5f) * 29.0f);
            atomicAdd(&my[bin], (t.w != -1) ? qw : 0u);
        }
    }
    __syncthreads();

    // ---- cross-lane reduction via WMMA: hist[n] = ones[16x32] @ P[32x16] ----
    // B-matrix (32x16 f16) layout used here: lanes 0-15 carry columns N=lane with
    // K=0..15; lanes 16-31 carry columns N=lane-16 with K=16..31. Since A is
    // all-ones, the K-slot ordering within a lane is irrelevant (pure column sum):
    // each column N must simply contain all 32 per-lane partials exactly once.
    const int n = lane & 15;
    const unsigned* src = &cnt[(wave * 32 + ((lane < 16) ? 0 : 16)) * LSTRIDE];

    v16h b0, b1;
    #pragma unroll
    for (int i = 0; i < 16; ++i) {
        // conflict-free: 16 consecutive words per half-wave, disjoint bank sets
        b0[i] = (_Float16)(float)src[i * LSTRIDE + n];        // bins 0..15
        b1[i] = (_Float16)(float)src[i * LSTRIDE + n + 16];   // bins 16..29 (+2 zero pads)
    }
    v16h a;
    #pragma unroll
    for (int i = 0; i < 16; ++i) a[i] = (_Float16)1.0f;

    v8f c0 = {};
    v8f c1 = {};
    // per-lane counts <= 64 -> exact in f16; f32 accumulate -> exact sums
    c0 = __builtin_amdgcn_wmma_f32_16x16x32_f16(false, a, false, b0, (short)0, c0, false, false);
    c1 = __builtin_amdgcn_wmma_f32_16x16x32_f16(false, a, false, b1, (short)0, c1, false, false);

    // D layout: VGPR0 = {lanes 0-15: M=0,N=lane | lanes 16-31: M=8,N=lane-16};
    // all D rows are identical (A all-ones), so c[0] is the histogram value.
    float* orow = out + (size_t)row * NBINS;
    if (lane < 16) {
        orow[lane] = c0[0];                 // bins 0..15
    } else if (lane < NBINS) {
        orow[lane] = c1[0];                 // bins 16..29
    }
}

extern "C" void kernel_launch(void* const* d_in, const int* in_sizes, int n_in,
                              void* d_out, int out_size, void* d_ws, size_t ws_size,
                              hipStream_t stream) {
    const float* simmat = (const float*)d_in[0];
    // d_in[1] = dlens: unused by the reference computation
    const int*   dtoks  = (const int*)d_in[2];
    const int*   qtoks  = (const int*)d_in[3];
    float*       out    = (float*)d_out;

    const int rows   = BB * CC * QQ;                 // 16384
    const int blocks = rows / WAVES_PER_BLOCK;       // 2048
    hipLaunchKernelGGL(CountHistogram_kernel, dim3(blocks), dim3(256), 0, stream,
                       simmat, dtoks, qtoks, out);
}
